// MultiHeadAttentionLayer_77129022702184
// MI455X (gfx1250) — matrix operand order repaired and checked by reading, not compiled, not run on hardware
//
#include <hip/hip_runtime.h>
#include <hip/hip_bf16.h>
#include <stddef.h>
#include <stdint.h>

// ---------------------------------------------------------------------------
// Multi-head attention for MI455X (gfx1250, wave32, WMMA + TDM).
//   B=4, S=2048, D=1024, H=16, Dh=64. fp32 I/O, bf16 WMMA compute, f32 accum.
//   K/V tiles staged to LDS with tensor_load_to_lds (TENSORcnt double buffer).
// ---------------------------------------------------------------------------

#define BD   4
#define SD   2048
#define DM   1024
#define NH   16
#define HD   64
#define NEG_INF_F (-1e9f)

typedef __attribute__((ext_vector_type(16))) __bf16 v16bf;
typedef __attribute__((ext_vector_type(8)))  float  v8f;
typedef __attribute__((ext_vector_type(4)))  unsigned int v4u;
typedef __attribute__((ext_vector_type(8)))  int  v8i;
typedef __attribute__((ext_vector_type(4)))  int  v4i;

__device__ __forceinline__ __bf16 f2bf(float f) { return (__bf16)f; }

// ---- WMMA fragment loaders (contiguous k = half*16 + s, not the ISA table order, 16-bit 16x16x32) --
// A (16x32, row-major src): lane = half*16+m holds row m; VGPR v element pair
//   k0 = half*16 + v*2
__device__ __forceinline__ v16bf load_a_f32(const float* __restrict__ src,
                                            int lda, int lane) {
    const int m = lane & 15, half = lane >> 4;
    v16bf a;
#pragma unroll
    for (int v = 0; v < 8; ++v) {
        const int k0 = half * 16 + 2 * v;
        a[2 * v]     = f2bf(src[m * lda + k0]);
        a[2 * v + 1] = f2bf(src[m * lda + k0 + 1]);
    }
    return a;
}

__device__ __forceinline__ v16bf load_a_bf16(const __bf16* __restrict__ src,
                                             int lda, int lane) {
    const int m = lane & 15, half = lane >> 4;
    v16bf a;
#pragma unroll
    for (int v = 0; v < 8; ++v) {
        const int k0 = half * 16 + 2 * v;
        a[2 * v]     = src[m * lda + k0];
        a[2 * v + 1] = src[m * lda + k0 + 1];
    }
    return a;
}

// B (32x16): lane = half*16+n holds column n; lanes 0-15 cover K=0..15,
// lanes 16-31 cover K=16..31; element (k,n) at src[k*ldb + n]
__device__ __forceinline__ v16bf load_b_f32(const float* __restrict__ src,
                                            int ldb, int lane) {
    const int n = lane & 15, half = lane >> 4;
    v16bf b;
#pragma unroll
    for (int v = 0; v < 8; ++v) {
        const int k0 = half * 16 + 2 * v;
        b[2 * v]     = f2bf(src[k0 * ldb + n]);
        b[2 * v + 1] = f2bf(src[(k0 + 1) * ldb + n]);
    }
    return b;
}

__device__ __forceinline__ v16bf load_b_bf16(const __bf16* __restrict__ src,
                                             int ldb, int lane) {
    const int n = lane & 15, half = lane >> 4;
    v16bf b;
#pragma unroll
    for (int v = 0; v < 8; ++v) {
        const int k0 = half * 16 + 2 * v;
        b[2 * v]     = src[k0 * ldb + n];
        b[2 * v + 1] = src[(k0 + 1) * ldb + n];
    }
    return b;
}

// B from transposed source: B[k][n] = src[n*ld + k]  (for Q*K^T)
__device__ __forceinline__ v16bf load_bT_bf16(const __bf16* __restrict__ src,
                                              int ld, int lane) {
    const int n = lane & 15, half = lane >> 4;
    v16bf b;
#pragma unroll
    for (int v = 0; v < 8; ++v) {
        const int k0 = half * 16 + 2 * v;
        b[2 * v]     = src[n * ld + k0];
        b[2 * v + 1] = src[n * ld + k0 + 1];
    }
    return b;
}

__device__ __forceinline__ v8f wmma_bf16(v16bf a, v16bf b, v8f c) {
    return __builtin_amdgcn_wmma_f32_16x16x32_bf16(
        /*neg_a=*/false, a, /*neg_b=*/false, b,
        /*c_mod=*/(short)0, c, /*reuse_a=*/false, /*reuse_b=*/false);
}

// ---------------------------------------------------------------------------
// TDM: async 2D tile DMA  global(bf16, row-major) -> LDS.
// D# packing per CDNA5 ISA 8.3/8.4:
//   g0: count=1 | lds_addr | global_addr[56:0] | type=2
//   g1: data_size=1(2B), tensor_dim0=row_elems, tensor_dim1=rows,
//       tile_dim0=row_elems, tile_dim1=rows, tensor_dim0_stride=stride
// This toolchain's builtin takes 6 args (g0, g1, g2, g3, extra v8i, cpol).
// ---------------------------------------------------------------------------
__device__ __forceinline__ void tdm_load_2d_bf16(const __bf16* gsrc,
                                                 unsigned lds_addr,
                                                 unsigned rows,
                                                 unsigned row_elems,
                                                 unsigned stride_elems) {
    const unsigned long long ga = (unsigned long long)(uintptr_t)gsrc;
    v4u g0;
    g0[0] = 1u;                                         // count=1, user desc
    g0[1] = lds_addr;                                   // LDS byte address
    g0[2] = (unsigned)(ga & 0xffffffffu);               // global_addr[31:0]
    g0[3] = (unsigned)((ga >> 32) & 0x01ffffffu) | (2u << 30);  // [56:32]|type=2
    v8i g1;
    g1[0] = (int)(1u << 16);                            // data_size = 2 bytes
    g1[1] = (int)((row_elems & 0xffffu) << 16);         // tensor_dim0 lo16
    g1[2] = (int)(((row_elems >> 16) & 0xffffu) |
                  ((rows & 0xffffu) << 16));            // td0 hi | td1 lo
    g1[3] = (int)(((rows >> 16) & 0xffffu) |
                  ((row_elems & 0xffffu) << 16));       // td1 hi | tile_dim0
    g1[4] = (int)(rows & 0xffffu);                      // tile_dim1 (tile_dim2=0)
    g1[5] = (int)stride_elems;                          // tensor_dim0_stride lo32
    g1[6] = 0;                                          // stride hi | td1_stride lo
    g1[7] = 0;
    v4i g2 = {0, 0, 0, 0};
    v4i g3 = {0, 0, 0, 0};
    v8i g4 = {0, 0, 0, 0, 0, 0, 0, 0};
    __builtin_amdgcn_tensor_load_to_lds(g0, g1, g2, g3, g4, 0);
}

// ---------------------------------------------------------------------------
// Kernel 1: fused QKV projection. z selects Q/K/V.  X[8192,1024] @ W + b,
// output head-split bf16 [B,H,S,Dh].  One wave -> one 16x16 tile.
// ---------------------------------------------------------------------------
__global__ __launch_bounds__(32) void mha_qkv_proj(
    const float* __restrict__ q_in, const float* __restrict__ k_in,
    const float* __restrict__ v_in,
    const float* __restrict__ Wq, const float* __restrict__ Wk,
    const float* __restrict__ Wv,
    const float* __restrict__ bq, const float* __restrict__ bk,
    const float* __restrict__ bv,
    __bf16* __restrict__ q_out, __bf16* __restrict__ k_out,
    __bf16* __restrict__ v_out) {
    const int lane = threadIdx.x;
    const int m0 = blockIdx.x * 16;            // token tile (B*S)/16
    const int n0 = blockIdx.y * 16;            // model-dim tile
    const int which = blockIdx.z;

    const float* X    = (which == 0) ? q_in : (which == 1) ? k_in : v_in;
    const float* W    = (which == 0) ? Wq   : (which == 1) ? Wk   : Wv;
    const float* bias = (which == 0) ? bq   : (which == 1) ? bk   : bv;
    __bf16*      out  = (which == 0) ? q_out : (which == 1) ? k_out : v_out;

    v8f acc = {};
    for (int k0 = 0; k0 < DM; k0 += 32) {
        if (k0 + 32 < DM) {
            __builtin_prefetch(W + (size_t)(k0 + 32) * DM + n0, 0, 1);
            __builtin_prefetch(X + (size_t)m0 * DM + (k0 + 32), 0, 1);
        }
        v16bf a = load_a_f32(X + (size_t)m0 * DM + k0, DM, lane);
        v16bf b = load_b_f32(W + (size_t)k0 * DM + n0, DM, lane);
        acc = wmma_bf16(a, b, acc);
    }

    // C/D layout: lane half*16+n, VGPR i -> (m = half*8+i, n)
    const int n = lane & 15, halfm = lane >> 4;
    const int col = n0 + n;
    const int h = col >> 6, d = col & 63;
    const float bv_ = bias[col];
#pragma unroll
    for (int i = 0; i < 8; ++i) {
        const int m = m0 + halfm * 8 + i;       // global token index
        const int bb = m >> 11, s = m & (SD - 1);
        out[(((size_t)bb * NH + h) * SD + s) * HD + d] = f2bf(acc[i] + bv_);
    }
}

// ---------------------------------------------------------------------------
// Kernel 2: flash attention.  One wave handles 16 queries x Dh=64 for one
// (b,h).  K/V streamed in 32-key chunks via TDM into double-buffered LDS,
// overlapped with compute through TENSORcnt.
// ---------------------------------------------------------------------------
__global__ __launch_bounds__(32) void mha_flash_attn(
    const __bf16* __restrict__ Q, const __bf16* __restrict__ K,
    const __bf16* __restrict__ V, const int* __restrict__ mask,
    __bf16* __restrict__ ctx) {
    __shared__ __bf16 k_lds[2][32 * HD];        // 2 x 4KB
    __shared__ __bf16 v_lds[2][32 * HD];        // 2 x 4KB
    __shared__ float  p_lds[16 * 32];           // P re-layout staging

    const int lane = threadIdx.x;
    const int q0 = blockIdx.x * 16;             // query tile
    const int h  = blockIdx.y;
    const int b  = blockIdx.z;

    const size_t head_off = (((size_t)b * NH + h) * SD) * HD;
    const __bf16* Qb = Q + head_off;
    const __bf16* Kb = K + head_off;
    const __bf16* Vb = V + head_off;
    const int*    Mb = mask + (size_t)b * SD * SD;

    const unsigned k_addr[2] = {(unsigned)(uintptr_t)&k_lds[0][0],
                                (unsigned)(uintptr_t)&k_lds[1][0]};
    const unsigned v_addr[2] = {(unsigned)(uintptr_t)&v_lds[0][0],
                                (unsigned)(uintptr_t)&v_lds[1][0]};

    // Q fragments: 16x64 -> two 16x32 A-frags
    const v16bf qa0 = load_a_bf16(Qb + (size_t)q0 * HD + 0,  HD, lane);
    const v16bf qa1 = load_a_bf16(Qb + (size_t)q0 * HD + 32, HD, lane);

    v8f o[4] = {v8f{}, v8f{}, v8f{}, v8f{}};
    float mrow[8], lrow[8];
#pragma unroll
    for (int i = 0; i < 8; ++i) { mrow[i] = -3.0e38f; lrow[i] = 0.0f; }

    const int n = lane & 15, halfm = lane >> 4;
    const float scale = 0.125f;                 // 1/sqrt(64)

    // prologue: DMA chunk 0 (K then V) into buffer 0
    tdm_load_2d_bf16(Kb, k_addr[0], 32, HD, HD);
    tdm_load_2d_bf16(Vb, v_addr[0], 32, HD, HD);

    for (int j = 0; j < SD; j += 32) {
        const int buf = (j >> 5) & 1;
        if (j + 32 < SD) {
            // issue next chunk into other buffer, then wait for current chunk
            tdm_load_2d_bf16(Kb + (size_t)(j + 32) * HD, k_addr[buf ^ 1], 32, HD, HD);
            tdm_load_2d_bf16(Vb + (size_t)(j + 32) * HD, v_addr[buf ^ 1], 32, HD, HD);
            __builtin_amdgcn_s_wait_tensorcnt(2);   // oldest 2 (this chunk) done
        } else {
            __builtin_amdgcn_s_wait_tensorcnt(0);
        }
        asm volatile("" ::: "memory");          // keep LDS reads below the wait

        const __bf16* Kc = &k_lds[buf][0];
        const __bf16* Vc = &v_lds[buf][0];

        // ---- scores: two 16x16 tiles over keys [j, j+16) and [j+16, j+32)
        v8f s0 = {}, s1 = {};
        s0 = wmma_bf16(qa0, load_bT_bf16(Kc + 0,            HD, lane), s0);
        s0 = wmma_bf16(qa1, load_bT_bf16(Kc + 32,           HD, lane), s0);
        s1 = wmma_bf16(qa0, load_bT_bf16(Kc + 16 * HD,      HD, lane), s1);
        s1 = wmma_bf16(qa1, load_bT_bf16(Kc + 16 * HD + 32, HD, lane), s1);

        // ---- scale + mask
#pragma unroll
        for (int i = 0; i < 8; ++i) {
            const int qrow = q0 + halfm * 8 + i;
            const int* mr = Mb + (size_t)qrow * SD + j;
            float v0 = s0[i] * scale;
            float v1 = s1[i] * scale;
            if (mr[n] == 0)      v0 = NEG_INF_F;
            if (mr[n + 16] == 0) v1 = NEG_INF_F;
            s0[i] = v0; s1[i] = v1;
        }

        // ---- online softmax (row reductions across 16-lane halves)
#pragma unroll
        for (int i = 0; i < 8; ++i) {
            float t = fmaxf(s0[i], s1[i]);
#pragma unroll
            for (int off = 8; off >= 1; off >>= 1)
                t = fmaxf(t, __shfl_xor(t, off, 16));
            const float mnew = fmaxf(mrow[i], t);
            const float p0 = __expf(s0[i] - mnew);
            const float p1 = __expf(s1[i] - mnew);
            const float corr = __expf(mrow[i] - mnew);
            float rsum = p0 + p1;
#pragma unroll
            for (int off = 8; off >= 1; off >>= 1)
                rsum += __shfl_xor(rsum, off, 16);
            lrow[i] = lrow[i] * corr + rsum;
            mrow[i] = mnew;
            s0[i] = p0; s1[i] = p1;
            o[0][i] *= corr; o[1][i] *= corr; o[2][i] *= corr; o[3][i] *= corr;
        }

        // ---- re-layout P (C-frag) -> A-frag through LDS
#pragma unroll
        for (int i = 0; i < 8; ++i) {
            const int r = halfm * 8 + i;
            p_lds[r * 32 + n]      = s0[i];
            p_lds[r * 32 + n + 16] = s1[i];
        }
        __syncthreads();                        // single wave: in-order LDS
        const v16bf pa = load_a_f32(p_lds, 32, lane);
        __syncthreads();

        // ---- O += P(16x32) * V(32x64)
#pragma unroll
        for (int t = 0; t < 4; ++t) {
            v16bf vb = load_b_bf16(Vc + t * 16, HD, lane);
            o[t] = wmma_bf16(pa, vb, o[t]);
        }
    }

    // ---- normalize and write context [B,S,D] (heads merged), bf16
#pragma unroll
    for (int i = 0; i < 8; ++i) {
        const float inv = 1.0f / lrow[i];
        const int r = q0 + halfm * 8 + i;
        const size_t base = ((size_t)b * SD + r) * DM + h * HD;
#pragma unroll
        for (int t = 0; t < 4; ++t)
            ctx[base + t * 16 + n] = f2bf(o[t][i] * inv);
    }
}

// ---------------------------------------------------------------------------
// Kernel 3: output projection. ctx(bf16)[8192,1024] @ Wo + bo -> f32 out.
// ---------------------------------------------------------------------------
__global__ __launch_bounds__(32) void mha_out_proj(
    const __bf16* __restrict__ ctx, const float* __restrict__ Wo,
    const float* __restrict__ bo, float* __restrict__ out) {
    const int lane = threadIdx.x;
    const int m0 = blockIdx.x * 16;
    const int n0 = blockIdx.y * 16;

    v8f acc = {};
    for (int k0 = 0; k0 < DM; k0 += 32) {
        if (k0 + 32 < DM)
            __builtin_prefetch(Wo + (size_t)(k0 + 32) * DM + n0, 0, 1);
        v16bf a = load_a_bf16(ctx + (size_t)m0 * DM + k0, DM, lane);
        v16bf b = load_b_f32(Wo + (size_t)k0 * DM + n0, DM, lane);
        acc = wmma_bf16(a, b, acc);
    }

    const int n = lane & 15, halfm = lane >> 4;
    const float bias = bo[n0 + n];
#pragma unroll
    for (int i = 0; i < 8; ++i) {
        const int m = m0 + halfm * 8 + i;
        out[(size_t)m * DM + n0 + n] = acc[i] + bias;
    }
}

// ---------------------------------------------------------------------------
extern "C" void kernel_launch(void* const* d_in, const int* in_sizes, int n_in,
                              void* d_out, int out_size, void* d_ws,
                              size_t ws_size, hipStream_t stream) {
    const float* query = (const float*)d_in[0];
    const float* key   = (const float*)d_in[1];
    const float* value = (const float*)d_in[2];
    const int*   mask  = (const int*)d_in[3];
    const float* Wq = (const float*)d_in[4];
    const float* bq = (const float*)d_in[5];
    const float* Wk = (const float*)d_in[6];
    const float* bk = (const float*)d_in[7];
    const float* Wv = (const float*)d_in[8];
    const float* bv = (const float*)d_in[9];
    const float* Wo = (const float*)d_in[10];
    const float* bo = (const float*)d_in[11];

    const size_t tensor_elems = (size_t)BD * SD * DM;   // 8M elements
    __bf16* q_ws = (__bf16*)d_ws;
    __bf16* k_ws = q_ws + tensor_elems;
    __bf16* v_ws = k_ws + tensor_elems;
    __bf16* ctx  = v_ws + tensor_elems;                 // 4 x 16 MB total

    // 1) QKV projections (z = which of Q/K/V)
    dim3 g1((BD * SD) / 16, DM / 16, 3);
    mha_qkv_proj<<<g1, 32, 0, stream>>>(query, key, value, Wq, Wk, Wv,
                                        bq, bk, bv, q_ws, k_ws, v_ws);

    // 2) flash attention per (q-tile, head, batch)
    dim3 g2(SD / 16, NH, BD);
    mha_flash_attn<<<g2, 32, 0, stream>>>(q_ws, k_ws, v_ws, mask, ctx);

    // 3) output projection
    dim3 g3((BD * SD) / 16, DM / 16);
    mha_out_proj<<<g3, 32, 0, stream>>>(ctx, Wo, bo, (float*)d_out);
}
